// CCEncoderVarDepth_27075473834048
// MI455X (gfx1250) — matrix operand order repaired and checked by reading, not compiled, not run on hardware
//
#include <hip/hip_runtime.h>
#include <hip/hip_bf16.h>

// ---------------------------------------------------------------------------
// Types for CDNA5 WMMA (wave32): A/B = v16 bf16, C/D = v8 f32
// ---------------------------------------------------------------------------
typedef __attribute__((ext_vector_type(16))) __bf16 v16bf;
typedef __attribute__((ext_vector_type(8)))  __bf16 v8bf;
typedef __attribute__((ext_vector_type(4)))  __bf16 v4bf;
typedef __attribute__((ext_vector_type(8)))  float  v8f;

#define WMMA_BF16(a, b, c)                                                   \
    __builtin_amdgcn_wmma_f32_16x16x32_bf16(false, (a), false, (b),          \
                                            (short)0, (c), false, false)

// ---- optional CDNA5 async global->LDS path (probe-guarded) ----------------
#if defined(__has_builtin)
#if __has_builtin(__builtin_amdgcn_global_load_async_to_lds_b128)
#define HAS_ASYNC_LDS 1
#endif
#endif
#ifndef HAS_ASYNC_LDS
#define HAS_ASYNC_LDS 0
#endif

#if HAS_ASYNC_LDS
// builtin signature (from hipcc diagnostic): params are pointers to
// "int __attribute__((vector_size(16)))" in AS1 (global) / AS3 (LDS).
typedef int v4i_vs __attribute__((vector_size(16)));
typedef __attribute__((address_space(1))) v4i_vs as1_v4i;
typedef __attribute__((address_space(3))) v4i_vs as3_v4i;
// generic->AS1: full 64-bit address; generic->AS3: low 32 bits = LDS offset
#define GPTR(p) ((as1_v4i*)(unsigned long long)(p))
#define LPTR(p) ((as3_v4i*)(unsigned long long)(unsigned)(unsigned long long)(p))
#endif

__device__ __forceinline__ void wait_async_lds() {
#if HAS_ASYNC_LDS
#if __has_builtin(__builtin_amdgcn_s_wait_asynccnt)
    __builtin_amdgcn_s_wait_asynccnt(0);
#else
    asm volatile("s_wait_asynccnt 0x0" ::: "memory");
#endif
#endif
}

__device__ __forceinline__ float silu_f(float x) {
    return x * (1.0f / (1.0f + __expf(-x)));
}

// fragment load: 16 contiguous bf16 in LDS -> two b128 reads
__device__ __forceinline__ v16bf ld_frag16(const __bf16* p) {
    v8bf lo = *(const v8bf*)p;
    v8bf hi = *(const v8bf*)(p + 8);
    v16bf r;
#pragma unroll
    for (int e = 0; e < 8; ++e) { r[e] = lo[e]; r[e + 8] = hi[e]; }
    return r;
}

// ---------------------------------------------------------------------------
// fp32 -> bf16 weight conversion. HWIO (3,3,Cin,Cout) flattened row-major is
// exactly the (K=9*Cin) x (N=Cout) row-major B matrix for implicit GEMM.
// ---------------------------------------------------------------------------
__global__ void cvt_f32_bf16_kernel(const float* __restrict__ in,
                                    __bf16* __restrict__ out, int n) {
    int i = blockIdx.x * blockDim.x + threadIdx.x;
    if (i < n) out[i] = (__bf16)in[i];
}

// ---------------------------------------------------------------------------
// Layer 0: Cin=1, 3x3 stride 2, fused SiLU, fp32 NCHW in -> bf16 NHWC out.
// ---------------------------------------------------------------------------
__global__ void conv0_kernel(const float* __restrict__ x,
                             const float* __restrict__ w,
                             __bf16* __restrict__ out,
                             int Hin, int Win, int Hout, int Wout) {
    int idx = blockIdx.x * blockDim.x + threadIdx.x;
    int total = 8 * Hout * Wout * 32;
    if (idx >= total) return;
    int co = idx & 31;
    int p  = idx >> 5;
    int wo = p % Wout;
    int t  = p / Wout;
    int ho = t % Hout;
    int n  = t / Hout;
    float s = 0.f;
#pragma unroll
    for (int kh = 0; kh < 3; ++kh) {
        int hi = ho * 2 + kh - 1;
        if ((unsigned)hi >= (unsigned)Hin) continue;
#pragma unroll
        for (int kw = 0; kw < 3; ++kw) {
            int wi = wo * 2 + kw - 1;
            if ((unsigned)wi >= (unsigned)Win) continue;
            s += x[((size_t)n * Hin + hi) * Win + wi] * w[(kh * 3 + kw) * 32 + co];
        }
    }
    out[idx] = (__bf16)silu_f(s);
}

// ---------------------------------------------------------------------------
// Small implicit-GEMM conv (Cout == 32): macro tile 64M x 32N, 8 waves,
// one 16x16 WMMA tile per wave. Fused SiLU -> bf16 NHWC. (Layers 1,2 are
// HBM-bandwidth-bound; WMMA density is irrelevant there.)
// ---------------------------------------------------------------------------
__global__ __launch_bounds__(256) void conv_wmma_small_kernel(
    const __bf16* __restrict__ in, const __bf16* __restrict__ wB,
    __bf16* __restrict__ outp,
    int Hin, int Win, int Cin, int Hout, int Wout, int Cout, int stride) {

    __shared__ __bf16 As[64][40];
    __shared__ __bf16 Bs[32][40];

    const int tid  = threadIdx.x;
    const int lane = tid & 31;
    const int wave = tid >> 5;
    const int wm   = wave & 3;
    const int wn   = wave >> 2;
    const int l16  = lane & 15;
    const int hi2  = (lane >> 4) & 1;

    const int mBlock = blockIdx.x * 64;
    const int nBlock = blockIdx.y * 32;
    const int K      = 9 * Cin;

    const int ar = tid >> 2;
    const int ac = (tid & 3) * 8;
    int p  = mBlock + ar;
    int wo = p % Wout;
    int t  = p / Wout;
    int ho = t % Hout;
    int n  = t / Hout;

    const int kr = tid >> 3;
    const int bc = (tid & 7) * 4;

    v8f acc = {0.f, 0.f, 0.f, 0.f, 0.f, 0.f, 0.f, 0.f};

    for (int kb = 0; kb < K; kb += 32) {
        int khkw = kb / Cin;
        int kh   = khkw / 3;
        int kw   = khkw % 3;
        int ci   = (kb % Cin) + ac;
        int hiy  = ho * stride + kh - 1;
        int wix  = wo * stride + kw - 1;
        v8bf av;
#pragma unroll
        for (int e = 0; e < 8; ++e) av[e] = (__bf16)0.f;
        if ((unsigned)hiy < (unsigned)Hin && (unsigned)wix < (unsigned)Win) {
            av = *(const v8bf*)&in[(((size_t)n * Hin + hiy) * Win + wix) * Cin + ci];
        }
        *(v8bf*)&As[ar][ac] = av;
        *(v4bf*)&Bs[kr][bc] =
            *(const v4bf*)&wB[(size_t)(kb + kr) * Cout + nBlock + bc];

        __syncthreads();

        v16bf a, b;
        const int arow = wm * 16 + l16;
        const int bcol = wn * 16 + l16;
#pragma unroll
        for (int e = 0; e < 16; ++e) {
            int ka  = (e & 7) + ((e >> 3) << 4) + hi2 * 8;
            a[e] = As[arow][ka];
            int kx  = (e & 7) + ((e >> 3) << 4) + hi2 * 8;
            b[e] = Bs[kx][bcol];
        }
        acc = WMMA_BF16(a, b, acc);
        __syncthreads();
    }

    const int co = nBlock + wn * 16 + l16;
#pragma unroll
    for (int r = 0; r < 8; ++r) {
        int P = mBlock + wm * 16 + hi2 * 8 + r;
        outp[(size_t)P * Cout + co] = (__bf16)silu_f(acc[r]);
    }
}

// ---------------------------------------------------------------------------
// Big implicit-GEMM conv (Cout >= 64): macro tile 128M x 64N, 8 waves (4x2),
// each wave computes 32x32 = 4 WMMA tiles per 32-K step. A tile is staged in
// fragment-ready (contiguous k = 16*hi2 + s) layout so fragments are b128 LDS reads,
// and staged via GLOBAL_LOAD_ASYNC_TO_LDS_B128 with double buffering when
// the toolchain exposes the builtin.
// MODE 0: fused SiLU -> bf16 | MODE 1: f32 + BN stats | MODE 2: f32 only
// ---------------------------------------------------------------------------
template <int MODE>
__global__ __launch_bounds__(256) void conv_wmma_big_kernel(
    const __bf16* __restrict__ in, const __bf16* __restrict__ wB,
    void* __restrict__ outp, float* __restrict__ stats,
    int Hin, int Win, int Cin, int Hout, int Wout, int Cout, int stride) {

    constexpr int PITCH = 40;  // 80B rows: 16B aligned, bank-staggered
    __shared__ __bf16 As[2][128][PITCH];
    __shared__ __bf16 Bt[2][64][PITCH];  // transposed: [cout][k]

    const int tid  = threadIdx.x;
    const int lane = tid & 31;
    const int wave = tid >> 5;
    const int wm   = wave & 3;   // 4 waves along M
    const int wn   = wave >> 2;  // 2 waves along N
    const int l16  = lane & 15;
    const int hi2  = (lane >> 4) & 1;

    const int mBlock = blockIdx.x * 128;
    const int nBlock = blockIdx.y * 64;
    const int K      = 9 * Cin;

    // A staging: row 0..127, half 0/1 -> 16 contiguous K (= ci) values.
    const int arow = tid >> 1;
    const int half = tid & 1;
    // K 8-blocks {0,1,2,3} -> storage cols {0,1,2,3}*8 (contiguous)
    const int sc0 = (half == 0) ? 0 : 16;   // (2*half)   * 8
    const int sc1 = (half == 0) ? 8 : 24;   // (2*half+1) * 8
    int p  = mBlock + arow;
    int wo = p % Wout;
    int t  = p / Wout;
    int ho = t % Hout;
    int n  = t / Hout;

    // B staging: 32 K rows x 64 couts; thread -> (kr, 8 couts)
    const int kr = tid >> 3;
    const int nc = (tid & 7) * 8;

    auto stageA = [&](int kb, int bsel) {
        int khkw = kb / Cin;  // Cin%32==0 -> one (kh,kw) tap per 32-K block
        int kh   = khkw / 3;
        int kw   = khkw % 3;
        int ci   = (kb % Cin) + half * 16;
        int hiy  = ho * stride + kh - 1;
        int wix  = wo * stride + kw - 1;
        __bf16* d0 = &As[bsel][arow][sc0];
        __bf16* d1 = &As[bsel][arow][sc1];
        if ((unsigned)hiy < (unsigned)Hin && (unsigned)wix < (unsigned)Win) {
            const __bf16* g =
                &in[(((size_t)n * Hin + hiy) * Win + wix) * Cin + ci];
#if HAS_ASYNC_LDS
            __builtin_amdgcn_global_load_async_to_lds_b128(GPTR(g),     LPTR(d0), 0, 0);
            __builtin_amdgcn_global_load_async_to_lds_b128(GPTR(g + 8), LPTR(d1), 0, 0);
#else
            *(v8bf*)d0 = *(const v8bf*)g;
            *(v8bf*)d1 = *(const v8bf*)(g + 8);
#endif
        } else {
            v8bf z;
#pragma unroll
            for (int e = 0; e < 8; ++e) z[e] = (__bf16)0.f;
            *(v8bf*)d0 = z;
            *(v8bf*)d1 = z;
        }
    };

    auto stageB = [&](int kb, int bsel) {
        v8bf v = *(const v8bf*)&wB[(size_t)(kb + kr) * Cout + nBlock + nc];
#pragma unroll
        for (int j = 0; j < 8; ++j) Bt[bsel][nc + j][kr] = v[j];
        if (kb + 32 < K)
            __builtin_prefetch(&wB[(size_t)(kb + 32 + kr) * Cout + nBlock + nc], 0, 3);
    };

    v8f acc[2][2];
#pragma unroll
    for (int i = 0; i < 2; ++i)
#pragma unroll
        for (int j = 0; j < 2; ++j)
            acc[i][j] = (v8f){0.f, 0.f, 0.f, 0.f, 0.f, 0.f, 0.f, 0.f};

    auto compute = [&](int bsel) {
        v16bf af[2], bf[2];
#pragma unroll
        for (int s = 0; s < 2; ++s) {
            af[s] = ld_frag16(&As[bsel][wm * 32 + s * 16 + l16][hi2 * 16]);
            bf[s] = ld_frag16(&Bt[bsel][wn * 32 + s * 16 + l16][hi2 * 16]);
        }
#pragma unroll
        for (int sm = 0; sm < 2; ++sm)
#pragma unroll
            for (int sn = 0; sn < 2; ++sn)
                acc[sm][sn] = WMMA_BF16(af[sm], bf[sn], acc[sm][sn]);
    };

    // pipelined main loop: stage(next) overlaps compute(cur)
    stageA(0, 0);
    stageB(0, 0);
    wait_async_lds();
    __syncthreads();
    int bsel = 0;
    for (int kb = 32; kb < K; kb += 32) {
        stageA(kb, bsel ^ 1);
        stageB(kb, bsel ^ 1);
        compute(bsel);
        wait_async_lds();
        __syncthreads();
        bsel ^= 1;
    }
    compute(bsel);

    // store per C/D layout: VGPR r -> M = r + hi2*8, N = l16
    const int coB = nBlock + wn * 32;
    const int pB  = mBlock + wm * 32;
#pragma unroll
    for (int sm = 0; sm < 2; ++sm) {
#pragma unroll
        for (int sn = 0; sn < 2; ++sn) {
            int co = coB + sn * 16 + l16;
#pragma unroll
            for (int r = 0; r < 8; ++r) {
                int P = pB + sm * 16 + hi2 * 8 + r;
                float v = acc[sm][sn][r];
                if constexpr (MODE == 0) {
                    ((__bf16*)outp)[(size_t)P * Cout + co] = (__bf16)silu_f(v);
                } else {
                    ((float*)outp)[(size_t)P * Cout + co] = v;
                }
            }
        }
    }
    if constexpr (MODE == 1) {
#pragma unroll
        for (int sn = 0; sn < 2; ++sn) {
            int c = coB + sn * 16 + l16;
            float s = 0.f, s2 = 0.f;
#pragma unroll
            for (int sm = 0; sm < 2; ++sm)
#pragma unroll
                for (int r = 0; r < 8; ++r) {
                    float v = acc[sm][sn][r];
                    s += v;
                    s2 += v * v;
                }
            atomicAdd(&stats[c], s);
            atomicAdd(&stats[Cout + c], s2);
        }
    }
}

// ---------------------------------------------------------------------------
// BN (per-channel over N*H*W) + SiLU, f32 -> bf16 NHWC.
// ---------------------------------------------------------------------------
__global__ void bn_silu_kernel(const float* __restrict__ xin,
                               const float* __restrict__ stats,
                               const float* __restrict__ gamma,
                               const float* __restrict__ beta,
                               __bf16* __restrict__ out,
                               int Cout, int total, float invCnt) {
    int i = blockIdx.x * blockDim.x + threadIdx.x;
    if (i >= total) return;
    int c = i % Cout;
    float mean = stats[c] * invCnt;
    float var  = stats[Cout + c] * invCnt - mean * mean;
    float sc   = gamma[c] * rsqrtf(var + 1e-5f);
    float y    = (xin[i] - mean) * sc + beta[c];
    out[i] = (__bf16)silu_f(y);
}

__global__ void zero_kernel(float* __restrict__ p, int n) {
    int i = blockIdx.x * blockDim.x + threadIdx.x;
    if (i < n) p[i] = 0.f;
}

__global__ void pool_kernel(const float* __restrict__ xin,
                            float* __restrict__ feats, int HW, int C) {
    int i = blockIdx.x * blockDim.x + threadIdx.x;
    if (i >= 8 * C) return;
    int c = i % C;
    int n = i / C;
    float s = 0.f;
    for (int sp = 0; sp < HW; ++sp)
        s += xin[((size_t)n * HW + sp) * C + c];
    feats[i] = s / (float)HW;
}

__global__ void fc_kernel(const float* __restrict__ feats,
                          const float* __restrict__ w,
                          const float* __restrict__ b,
                          float* __restrict__ out) {
    int i = blockIdx.x * blockDim.x + threadIdx.x;
    if (i >= 8 * 128) return;
    int o = i % 128;
    int n = i / 128;
    float s = b[o];
    for (int c = 0; c < 1024; ++c)
        s += feats[n * 1024 + c] * w[o * 1024 + c];
    out[i] = s;
}

// ---------------------------------------------------------------------------
// Host-side orchestration
// ---------------------------------------------------------------------------
extern "C" void kernel_launch(void* const* d_in, const int* in_sizes, int n_in,
                              void* d_out, int out_size, void* d_ws, size_t ws_size,
                              hipStream_t stream) {
    struct Spec { int cin, cout, stride; };
    static const Spec S[18] = {
        {1, 32, 2},   {32, 32, 1},   {32, 32, 1},
        {32, 64, 2},  {64, 64, 1},   {64, 64, 1},
        {64, 128, 2}, {128, 128, 1}, {128, 128, 1},
        {128, 256, 2},{256, 256, 1}, {256, 256, 1},
        {256, 512, 2},{512, 512, 1}, {512, 512, 1},
        {512, 1024, 2},{1024, 1024, 1},{1024, 1024, 1}};

    int H[19], W[19];
    H[0] = 768; W[0] = 256;
    for (int i = 0; i < 18; ++i) {
        H[i + 1] = (S[i].stride == 2) ? H[i] / 2 : H[i];
        W[i + 1] = (S[i].stride == 2) ? W[i] / 2 : W[i];
    }

    char* ws = (char*)d_ws;
    size_t off = 0;
    auto carve = [&](size_t bytes) -> void* {
        void* r = ws + off;
        off += (bytes + 255) & ~(size_t)255;
        return r;
    };
    __bf16* wbf[18];
    for (int i = 0; i < 18; ++i)
        wbf[i] = (__bf16*)carve((size_t)9 * S[i].cin * S[i].cout * sizeof(__bf16));
    const size_t maxActElems = (size_t)8 * 384 * 128 * 32;
    __bf16* actA = (__bf16*)carve(maxActElems * sizeof(__bf16));
    __bf16* actB = (__bf16*)carve(maxActElems * sizeof(__bf16));
    float*  ftmp = (float*)carve((size_t)8 * 96 * 32 * 128 * sizeof(float));
    float*  stats = (float*)carve(2048 * sizeof(float));
    float*  feats = (float*)carve((size_t)8 * 1024 * sizeof(float));
    (void)ws_size; (void)n_in; (void)in_sizes; (void)out_size;

    for (int i = 0; i < 18; ++i) {
        int cnt = 9 * S[i].cin * S[i].cout;
        cvt_f32_bf16_kernel<<<(cnt + 255) / 256, 256, 0, stream>>>(
            (const float*)d_in[1 + i], wbf[i], cnt);
    }

    {
        int total = 8 * H[1] * W[1] * 32;
        conv0_kernel<<<(total + 255) / 256, 256, 0, stream>>>(
            (const float*)d_in[0], (const float*)d_in[1], actA,
            H[0], W[0], H[1], W[1]);
    }

    __bf16* cur = actA;
    __bf16* nxt = actB;
    for (int i = 1; i < 18; ++i) {
        int Hin = H[i], Win = W[i], Ho = H[i + 1], Wo = W[i + 1];
        int Cin = S[i].cin, Cout = S[i].cout, st = S[i].stride;
        int M = 8 * Ho * Wo;
        if (Cout == 32) {  // layers 1,2
            dim3 grid(M / 64, 1);
            conv_wmma_small_kernel<<<grid, 256, 0, stream>>>(
                cur, wbf[i], nxt, Hin, Win, Cin, Ho, Wo, Cout, st);
            __bf16* tsw = cur; cur = nxt; nxt = tsw;
        } else {
            dim3 grid(M / 128, Cout / 64);
            if (i < 6) {
                conv_wmma_big_kernel<0><<<grid, 256, 0, stream>>>(
                    cur, wbf[i], (void*)nxt, nullptr,
                    Hin, Win, Cin, Ho, Wo, Cout, st);
                __bf16* tsw = cur; cur = nxt; nxt = tsw;
            } else if (i < 17) {
                int statN = 2 * Cout;
                zero_kernel<<<(statN + 255) / 256, 256, 0, stream>>>(stats, statN);
                conv_wmma_big_kernel<1><<<grid, 256, 0, stream>>>(
                    cur, wbf[i], (void*)ftmp, stats,
                    Hin, Win, Cin, Ho, Wo, Cout, st);
                int total = M * Cout;
                const float* g  = (const float*)d_in[19 + (i - 6)];
                const float* be = (const float*)d_in[30 + (i - 6)];
                bn_silu_kernel<<<(total + 255) / 256, 256, 0, stream>>>(
                    ftmp, stats, g, be, nxt, Cout, total, 1.0f / (float)M);
                __bf16* tsw = cur; cur = nxt; nxt = tsw;
            } else {
                conv_wmma_big_kernel<2><<<grid, 256, 0, stream>>>(
                    cur, wbf[17], (void*)ftmp, nullptr,
                    Hin, Win, Cin, Ho, Wo, Cout, st);
            }
        }
    }

    pool_kernel<<<(8 * 1024 + 255) / 256, 256, 0, stream>>>(
        ftmp, feats, H[18] * W[18], 1024);
    fc_kernel<<<(1024 + 255) / 256, 256, 0, stream>>>(
        feats, (const float*)d_in[41], (const float*)d_in[42], (float*)d_out);
}
